// CompressedLinear_65953517798089
// MI455X (gfx1250) — matrix operand order repaired and checked by reading, not compiled, not run on hardware
//
#include <hip/hip_runtime.h>

// CompressedLinear: out[M,N] = X[M,K] * (W_int8[N,K] * scale[N])^T + bias[N]
// M = 8192 (B*S), N = 4096, K = 4096.  Compute-bound GEMM -> f16 WMMA with
// hi/lo split of fp32 X for fp32-class accuracy; int8 weights are exact in f16.

#define M_TOT 8192
#define N_TOT 4096
#define K_TOT 4096
#define BM 128
#define BN 128
#define BK 32
#define LDA 40                  // BK + 8 halves of padding (bank-conflict relief)
#define NKS (K_TOT / BK)        // 128 k-steps

typedef __attribute__((ext_vector_type(16))) _Float16 v16h;
typedef __attribute__((ext_vector_type(8)))  _Float16 v8h;
typedef __attribute__((ext_vector_type(4)))  _Float16 v4h;
typedef __attribute__((ext_vector_type(8)))  float    v8f;
typedef __attribute__((ext_vector_type(4)))  float    v4f;
typedef __attribute__((ext_vector_type(4)))  int      v4i;

#if defined(__has_builtin) && __has_builtin(__builtin_amdgcn_sched_barrier)
#define SCHED_FENCE() __builtin_amdgcn_sched_barrier(0)
#else
#define SCHED_FENCE()
#endif

static __device__ __forceinline__ v16h cat16(v8h a, v8h b) {
    return __builtin_shufflevector(a, b, 0, 1, 2, 3, 4, 5, 6, 7,
                                         8, 9, 10, 11, 12, 13, 14, 15);
}

// Convert this thread's staged registers to f16 and store into LDS buffer `buf`.
#define STORE_TILES(buf)                                                      \
    do {                                                                      \
        _Pragma("unroll")                                                     \
        for (int i = 0; i < 4; ++i) {                                         \
            const int row = rbase + 32 * i;                                   \
            v4h hx, lx, hw;                                                   \
            _Pragma("unroll")                                                 \
            for (int j = 0; j < 4; ++j) {                                     \
                const float f = xr[i][j];                                     \
                const _Float16 h = (_Float16)f;                               \
                hx[j] = h;                                                    \
                lx[j] = (_Float16)(f - (float)h);  /* residual, exact */      \
                hw[j] = (_Float16)(short)wr[i][j]; /* int8 exact in f16 */    \
            }                                                                 \
            *(v4h*)&XsH[buf][row * LDA + c4] = hx;                            \
            *(v4h*)&XsL[buf][row * LDA + c4] = lx;                            \
            *(v4h*)&Wsh[buf][row * LDA + c4] = hw;                            \
        }                                                                     \
    } while (0)

__global__ __launch_bounds__(256)
void CompressedLinear_65953517798089_kernel(const float* __restrict__ x,
                                            const int*   __restrict__ w,
                                            const float* __restrict__ scale,
                                            const float* __restrict__ bias,
                                            float*       __restrict__ out) {
    __shared__ _Float16 XsH[2][BM * LDA];   // X hi halves   (2 x 10 KB)
    __shared__ _Float16 XsL[2][BM * LDA];   // X lo residual (2 x 10 KB)
    __shared__ _Float16 Wsh[2][BN * LDA];   // W as f16      (2 x 10 KB)

    const int tid  = (int)threadIdx.x;
    const int lane = tid & 31;              // wave32
    const int wave = tid >> 5;              // 0..7
    const int wm   = wave & 1;              // 2 M-groups of 64 rows
    const int wn   = wave >> 1;             // 4 N-groups of 32 cols
    const int grp  = lane >> 4;             // 0/1 half-wave
    const int l16  = lane & 15;

    const int m_base = (int)blockIdx.y * BM;
    const int n_base = (int)blockIdx.x * BN;

    // Cooperative load mapping: 256 threads x (4 strips x 4 elems) per tile.
    const int rbase = tid >> 3;             // 0..31 (strips repeat at +32)
    const int c4    = (tid & 7) * 4;        // k-column within BK

    // Loop-invariant per-lane BYTE offsets (fit in i32): the k-advance is a
    // uniform base-pointer bump, so loads match the saddr (GVS) pattern
    //   addr = SGPR_base64 + sext(VGPR_i32)   -> global_load ... saddr
    int xboff[4], wboff[4];
#pragma unroll
    for (int i = 0; i < 4; ++i) {
        xboff[i] = ((m_base + rbase + 32 * i) * K_TOT + c4) * 4;  // bytes
        wboff[i] = ((n_base + rbase + 32 * i) * K_TOT + c4) * 4;  // bytes
    }

    v8f acc[4][2];
#pragma unroll
    for (int mt = 0; mt < 4; ++mt)
#pragma unroll
        for (int nt = 0; nt < 2; ++nt)
            acc[mt][nt] = (v8f){0.f, 0.f, 0.f, 0.f, 0.f, 0.f, 0.f, 0.f};

    v4f xr[4];
    v4i wr[4];

    // Prologue: stage k-tile 0.
    {
        const char* __restrict__ xb = (const char*)x;
        const char* __restrict__ wb = (const char*)w;
#pragma unroll
        for (int i = 0; i < 4; ++i) {
            xr[i] = *(const v4f*)(xb + xboff[i]);
            wr[i] = *(const v4i*)(wb + wboff[i]);
        }
    }
    STORE_TILES(0);
    __syncthreads();

    for (int ks = 0; ks < NKS; ++ks) {
        const int cur = ks & 1;

        // Prefetch next k-tile: issued as one batch at the TOP of the
        // iteration; the sched fence keeps the scheduler from sinking the
        // loads below the WMMA chain, so global latency is fully overlapped.
        if (ks + 1 < NKS) {
            const char* __restrict__ xk = (const char*)(x + (ks + 1) * BK);
            const char* __restrict__ wk = (const char*)(w + (ks + 1) * BK);
#pragma unroll
            for (int i = 0; i < 4; ++i) {
                xr[i] = *(const v4f*)(xk + xboff[i]);
                wr[i] = *(const v4i*)(wk + wboff[i]);
            }
            SCHED_FENCE();
        }

        const _Float16* __restrict__ xh = XsH[cur];
        const _Float16* __restrict__ xl = XsL[cur];
        const _Float16* __restrict__ ws = Wsh[cur];

        // B fragments (32x16 f16): lane holds column l16, K-run grp*16..+15.
        v16h bfr[2];
#pragma unroll
        for (int nt = 0; nt < 2; ++nt) {
            const int n = wn * 32 + nt * 16 + l16;
            const v8h b0 = *(const v8h*)&ws[n * LDA + grp * 16];
            const v8h b1 = *(const v8h*)&ws[n * LDA + grp * 16 + 8];
            bfr[nt] = cat16(b0, b1);
        }

        // Two passes: pass 0 = X-hi, pass 1 = X-lo residual.  One A fragment
        // set live at a time; same-acc WMMAs are 8 ops apart (no hazard nops).
#pragma unroll
        for (int p = 0; p < 2; ++p) {
            const _Float16* __restrict__ xs = p ? xl : xh;
#pragma unroll
            for (int mt = 0; mt < 4; ++mt) {
                const int m = wm * 64 + mt * 16 + l16;
                const v8h a0 = *(const v8h*)&xs[m * LDA + grp * 16];
                const v8h a1 = *(const v8h*)&xs[m * LDA + grp * 16 + 8];
                const v16h af = cat16(a0, a1);
#pragma unroll
                for (int nt = 0; nt < 2; ++nt) {
                    acc[mt][nt] = __builtin_amdgcn_wmma_f32_16x16x32_f16(
                        false, af, false, bfr[nt], (short)0, acc[mt][nt],
                        false, false);
                }
            }
        }

        if (ks + 1 < NKS) {
            STORE_TILES(cur ^ 1);   // other buffer: safe since last read at ks-1
            __syncthreads();
        }
    }

    // Epilogue: D tile layout — VGPR r -> (m = r + 8*grp, n = l16).
#pragma unroll
    for (int nt = 0; nt < 2; ++nt) {
        const int n = n_base + wn * 32 + nt * 16 + l16;
        const float s  = scale[n];
        const float bb = bias[n];
#pragma unroll
        for (int mt = 0; mt < 4; ++mt) {
            const int mr = m_base + wm * 64 + mt * 16 + grp * 8;
#pragma unroll
            for (int r = 0; r < 8; ++r) {
                out[(size_t)(mr + r) * N_TOT + n] = acc[mt][nt][r] * s + bb;
            }
        }
    }
}

extern "C" void kernel_launch(void* const* d_in, const int* in_sizes, int n_in,
                              void* d_out, int out_size, void* d_ws, size_t ws_size,
                              hipStream_t stream) {
    const float* x     = (const float*)d_in[0];
    const int*   w     = (const int*)d_in[1];     // int8 values stored as int32
    const float* scale = (const float*)d_in[2];
    const float* bias  = (const float*)d_in[3];
    float*       out   = (float*)d_out;

    dim3 grid(N_TOT / BN, M_TOT / BM);            // 32 x 64 workgroups
    dim3 block(256);                              // 8 wave32s
    hipLaunchKernelGGL(CompressedLinear_65953517798089_kernel,
                       grid, block, 0, stream, x, w, scale, bias, out);
}
